// TestCUDAMoe_6605659701834
// MI455X (gfx1250) — matrix operand order repaired and checked by reading, not compiled, not run on hardware
//
#include <hip/hip_runtime.h>
#include <hip/hip_bf16.h>
#include <stdint.h>

#define HID   2048
#define NEXP  16
#define EW    512
#define NTOK  8192
#define CAP   8192
#define MTILE 32
#define NC    64      // phase-1 N-chunk / phase-2 H-chunk
#define KC    32      // K-chunk per WMMA stage

typedef __attribute__((ext_vector_type(16))) __bf16 v16bf;
typedef __attribute__((ext_vector_type(8)))  float  v8f;

union Frag { v16bf v; unsigned u[8]; };

// One-instruction packed f32->bf16 conversion (pins codegen, avoids v_perm fixups)
__device__ __forceinline__ unsigned pack2_bf16(float a, float b) {
    unsigned r;
    asm("v_cvt_pk_bf16_f32 %0, %1, %2" : "=v"(r) : "v"(a), "v"(b));
    return r;
}
__device__ __forceinline__ unsigned short f32_to_bf16(float f) {
    union { __bf16 h; unsigned short u; } r;
    r.h = (__bf16)f;
    return r.u;
}

// ---- CDNA5 async global->LDS DMA (ASYNCcnt path) --------------------------
// LDS address for the async op: generic LDS pointer's low 32 bits are the
// workgroup-relative LDS byte offset.
__device__ __forceinline__ unsigned lds_off(const void* p) {
    return (unsigned)(uintptr_t)p;
}
__device__ __forceinline__ void async_load_b128(unsigned ldsaddr, const float* gptr) {
    asm volatile("global_load_async_to_lds_b128 %0, %1, off"
                 :: "v"(ldsaddr), "v"((uint64_t)(uintptr_t)gptr)
                 : "memory");
}
__device__ __forceinline__ void wait_async0() {
    asm volatile("s_wait_asynccnt 0x0" ::: "memory");
}

// ---------------------------------------------------------------------------
// Kernel 0: zero output + expert counters
// ---------------------------------------------------------------------------
__global__ void zero_kernel(float* __restrict__ out, int n4, int* __restrict__ cnt) {
    int i = blockIdx.x * blockDim.x + threadIdx.x;
    if (i < n4) ((float4*)out)[i] = make_float4(0.f, 0.f, 0.f, 0.f);
    if (i < NEXP) cnt[i] = 0;
}

// ---------------------------------------------------------------------------
// Kernel 1: gating (logits -> top-2 renormalized -> per-expert lists)
// ---------------------------------------------------------------------------
__global__ __launch_bounds__(128) void gate_kernel(
    const float* __restrict__ x, const float* __restrict__ gw,
    int* __restrict__ cnt, int* __restrict__ tok, float* __restrict__ wt)
{
    int token = blockIdx.x * 128 + threadIdx.x;
    if (token >= NTOK) return;
    const float* xr = x + (size_t)token * HID;

    float acc[NEXP];
    #pragma unroll
    for (int e = 0; e < NEXP; ++e) acc[e] = 0.f;

    for (int h = 0; h < HID; h += 4) {
        float4 xv = *(const float4*)(xr + h);
        #pragma unroll
        for (int e = 0; e < NEXP; ++e) {
            float4 gv = *(const float4*)(gw + (size_t)e * HID + h);
            acc[e] += xv.x * gv.x + xv.y * gv.y + xv.z * gv.z + xv.w * gv.w;
        }
    }

    int i0 = 0; float v0 = acc[0];
    #pragma unroll
    for (int e = 1; e < NEXP; ++e) if (acc[e] > v0) { v0 = acc[e]; i0 = e; }
    int i1 = (i0 == 0) ? 1 : 0; float v1 = acc[i1];
    #pragma unroll
    for (int e = 0; e < NEXP; ++e)
        if (e != i0 && acc[e] > v1) { v1 = acc[e]; i1 = e; }

    // pair-softmax (global Z cancels); fast rcp is plenty accurate here
    float w0 = __builtin_amdgcn_rcpf(1.f + __expf(v1 - v0));
    float w1 = 1.f - w0;

    int s0 = atomicAdd(&cnt[i0], 1);
    if (s0 < CAP) { tok[i0 * CAP + s0] = token; wt[i0 * CAP + s0] = w0; }
    int s1 = atomicAdd(&cnt[i1], 1);
    if (s1 < CAP) { tok[i1 * CAP + s1] = token; wt[i1 * CAP + s1] = w1; }
}

// ---------------------------------------------------------------------------
// Kernel 2: per-expert fused  ir = (X@u_e^T)*silu(X@g_e^T) ;  out += w*(ir@d_e)
// grid = (CAP/MTILE, NEXP), block = 256 (8 waves); wave = (m = wv&1, nb = wv>>1)
// ---------------------------------------------------------------------------
__global__ __launch_bounds__(256) void moe_expert_kernel(
    const float* __restrict__ x, const float* __restrict__ u,
    const float* __restrict__ g, const float* __restrict__ d,
    const int* __restrict__ cnt, const int* __restrict__ tok,
    const float* __restrict__ wt, float* __restrict__ out)
{
    const int e    = blockIdx.y;
    const int tile = blockIdx.x;
    int nthis = cnt[e]; if (nthis > CAP) nthis = CAP;
    if (tile * MTILE >= nthis) return;

    // f32 staging tiles (async DMA targets), rows padded to 34 floats (8B-aligned pairs)
    __shared__ __attribute__((aligned(16))) float xs32[MTILE][34];   // X tile 32 x K32
    __shared__ __attribute__((aligned(16))) float us32[NC][34];      // u rows / d rows (phase 2)
    __shared__ __attribute__((aligned(16))) float gs32[NC][34];      // g rows
    __shared__ __attribute__((aligned(16))) unsigned short irs[MTILE][522]; // ir 32x512 bf16
    __shared__ int   tk_sh[MTILE];
    __shared__ float wt_sh[MTILE];

    const int tid  = threadIdx.x;
    const int lane = tid & 31;
    const int wv   = tid >> 5;
    const int m    = wv & 1;           // 16-row m-block
    const int nb   = wv >> 1;          // 16-col n-block within 64-wide chunk
    const int lm   = lane & 15;
    const bool hi  = lane >= 16;
    const int dselA = hi ? 4 : 0;      // A-frag dword select (packed bf16 source)
    const int moff  = hi ? 8 : 0;      // C/D row offset

    if (tid < MTILE) {
        int slot = tile * MTILE + tid;
        bool ok = slot < nthis;
        tk_sh[tid] = ok ? tok[e * CAP + slot] : 0;
        wt_sh[tid] = ok ? wt[e * CAP + slot] : 0.f;
    }
    __syncthreads();

    // staging slot for this thread: row r (8 lanes of b128 per 32-float row)
    const int sr = tid >> 3;           // 0..31
    const int sq = (tid & 7) * 4;      // float col 0,4,..,28

    // ---------------- Phase 1: ir tile (32 x 512), 8 chunks of 64 cols -------
    for (int nc = 0; nc < 8; ++nc) {
        v8f accU = {0,0,0,0,0,0,0,0};
        v8f accG = {0,0,0,0,0,0,0,0};

        for (int kk = 0; kk < HID; kk += KC) {
            __syncthreads();
            // X tile: 256 b128 slots (one per thread)
            async_load_b128(lds_off(&xs32[sr][sq]),
                            x + (size_t)tk_sh[sr] * HID + kk + sq);
            // u/g tiles: 64 rows x 8 slots = 512 slots each (2 per thread)
            #pragma unroll
            for (int i = 0; i < 2; ++i) {
                int slot = tid + i * 256;
                int r = slot >> 3, q = (slot & 7) * 4;
                size_t row = (size_t)(e * EW + nc * NC + r);
                async_load_b128(lds_off(&us32[r][q]), u + row * HID + kk + q);
                async_load_b128(lds_off(&gs32[r][q]), g + row * HID + kk + q);
            }
            wait_async0();
            __syncthreads();

            // A fragment (16x32): lane=row; VGPR j holds K pair (0-7/8-15/16-23/24-31 interleave)
            Frag a;
            const int rowA = m * 16 + lm;
            #pragma unroll
            for (int j = 0; j < 8; ++j) {
                int kA = (j < 4 ? 2 * j : 16 + 2 * (j - 4)) + (hi ? 8 : 0);
                float2 p = *(const float2*)&xs32[rowA][kA];
                a.u[j] = pack2_bf16(p.x, p.y);
            }
            // B fragments (32x16): lane=col; VGPR j holds K pair (0-7/8-15/16-23/24-31 interleave, same as A)
            Frag bu, bg;
            const int rowB = nb * 16 + lm;
            #pragma unroll
            for (int j = 0; j < 8; ++j) {
                int kB = (j < 4 ? 2 * j : 16 + 2 * (j - 4)) + (hi ? 8 : 0);
                float2 pu = *(const float2*)&us32[rowB][kB];
                float2 pg = *(const float2*)&gs32[rowB][kB];
                bu.u[j] = pack2_bf16(pu.x, pu.y);
                bg.u[j] = pack2_bf16(pg.x, pg.y);
            }
            accU = __builtin_amdgcn_wmma_f32_16x16x32_bf16(
                false, a.v, false, bu.v, (short)0, accU, false, false);
            accG = __builtin_amdgcn_wmma_f32_16x16x32_bf16(
                false, a.v, false, bg.v, (short)0, accG, false, false);
        }

        // silu(g)*u -> bf16 ir tile in LDS (disjoint region per wave)
        {
            int col = nc * NC + nb * 16 + lm;
            #pragma unroll
            for (int v = 0; v < 8; ++v) {
                float uv = accU[v];
                float gv = accG[v];
                float s  = __builtin_amdgcn_rcpf(1.f + __expf(-gv));   // v_rcp_f32
                float ir = uv * gv * s;
                irs[m * 16 + v + moff][col] = f32_to_bf16(ir);
            }
        }
    }
    __syncthreads();   // ir tile complete & visible

    // ---------------- Phase 2: out += w * (ir @ d_e) -------------------------
    const unsigned* irsd = (const unsigned*)&irs[0][0];   // row stride 261 dwords
    const size_t dstride = (size_t)(NEXP * EW);           // 8192
    const int rowA = m * 16 + lm;
    const int rowB = nb * 16 + lm;

    for (int hb = 0; hb < HID; hb += NC) {
        v8f acc = {0,0,0,0,0,0,0,0};

        for (int kk = 0; kk < EW; kk += KC) {
            __syncthreads();
            // stage d[hb+r][e*512+kk ..] : 64 rows x 32 f32 (2 b128 slots/thread)
            #pragma unroll
            for (int i = 0; i < 2; ++i) {
                int slot = tid + i * 256;
                int r = slot >> 3, q = (slot & 7) * 4;
                async_load_b128(lds_off(&us32[r][q]),
                                d + (size_t)(hb + r) * dstride + e * EW + kk + q);
            }
            wait_async0();
            __syncthreads();

            Frag a;
            #pragma unroll
            for (int j = 0; j < 8; ++j) {
                int dc = (j < 4 ? j : j + 4) + dselA;
                a.u[j] = irsd[rowA * 261 + (kk >> 1) + dc];
            }
            Frag b;
            #pragma unroll
            for (int j = 0; j < 8; ++j) {
                int kB = (j < 4 ? 2 * j : 16 + 2 * (j - 4)) + (hi ? 8 : 0);
                float2 p = *(const float2*)&us32[rowB][kB];
                b.u[j] = pack2_bf16(p.x, p.y);
            }
            acc = __builtin_amdgcn_wmma_f32_16x16x32_bf16(
                false, a.v, false, b.v, (short)0, acc, false, false);
        }

        // scale by gate weight and scatter-add
        int h = hb + nb * 16 + lm;
        #pragma unroll
        for (int v = 0; v < 8; ++v) {
            int mrow = m * 16 + v + moff;
            float val = acc[v] * wt_sh[mrow];
            atomicAdd(out + (size_t)tk_sh[mrow] * HID + h, val);
        }
    }
}

// ---------------------------------------------------------------------------
extern "C" void kernel_launch(void* const* d_in, const int* in_sizes, int n_in,
                              void* d_out, int out_size, void* d_ws, size_t ws_size,
                              hipStream_t stream) {
    (void)in_sizes; (void)n_in; (void)out_size; (void)ws_size;
    const float* x  = (const float*)d_in[0];   // hid (4,2048,2048)
    const float* gw = (const float*)d_in[1];   // gate_w (16,2048)
    const float* u  = (const float*)d_in[2];   // u (8192,2048)
    const float* g  = (const float*)d_in[3];   // g (8192,2048)
    const float* d  = (const float*)d_in[4];   // d (2048,8192)
    float* out = (float*)d_out;

    char* ws  = (char*)d_ws;
    int*   cnt = (int*)ws;                                   // 16 ints
    int*   tok = (int*)(ws + 256);                           // 16*8192 ints
    float* wt  = (float*)(ws + 256 + (size_t)NEXP * CAP * sizeof(int));

    const int n4 = NTOK * HID / 4;
    zero_kernel<<<(n4 + 255) / 256, 256, 0, stream>>>(out, n4, cnt);
    gate_kernel<<<NTOK / 128, 128, 0, stream>>>(x, gw, cnt, tok, wt);
    dim3 grid(CAP / MTILE, NEXP);
    moe_expert_kernel<<<grid, 256, 0, stream>>>(x, u, g, d, cnt, tok, wt, out);
}
